// ReAttentionTransformerBlock_6803228196955
// MI455X (gfx1250) — matrix operand order repaired and checked by reading, not compiled, not run on hardware
//
#include <hip/hip_runtime.h>
#include <hip/hip_bf16.h>
#include <stdint.h>

typedef _Float16 f16;
typedef __attribute__((ext_vector_type(16))) _Float16 v16h;
typedef __attribute__((ext_vector_type(8)))  float    v8f;

#define L_   5
#define B_   64
#define N_   256
#define CHN_ 3
#define P_   16
#define C_   768
#define HID_ 1536
#define H_   8
#define HD_  96
#define EPS_ 1e-5f
#define SCALE_ 0.10206207261596575f   // 96^-0.5

// ---------------------------------------------------------------------------
// Batched WMMA GEMM:  C[M,N](f32) = alpha * A[M,K](f16 row-major) x B
// B is ALWAYS stored as [N,K] row-major (element (k,n) at B[n*ldb+k]),
// so every global load is contiguous along K. Weights are pre-transposed.
// Block = 256 threads (8 waves, wave32). Block tile 128x64, K-step 32.
// Waves in a 4x2 grid; each wave computes a 32x32 sub-tile = 4 WMMAs/K-step.
// K-loop is software-pipelined with double-buffered LDS: global loads for
// step i+1 are issued before the WMMAs of step i; ONE barrier per K-step.
// Requires: M % 128 == 0, K % 32 == 0 and K >= 32 (true at all call sites);
// N guarded branchlessly (clamped row address + zero fill).
// ---------------------------------------------------------------------------
#define TM_ 128
#define TN_ 64
#define AST 34   // LDS row stride (f16 units), padded to dodge bank conflicts
#define BST 34

__global__ __launch_bounds__(256) void k_gemm(
    const f16* __restrict__ A, int lda, long long sA,
    const f16* __restrict__ Bm, int ldb, long long sB,
    float* __restrict__ C, int ldc, long long sC,
    int M, int N, int K, float alpha)
{
    __shared__ f16 la[2][TM_ * AST];
    __shared__ f16 lb[2][TN_ * BST];

    const int tid  = threadIdx.x;
    const int lane = tid & 31;
    const int w    = tid >> 5;
    const int m0   = blockIdx.y * TM_;
    const int n0   = blockIdx.x * TN_;
    const long long ab = (long long)blockIdx.z * sA;
    const long long bb = (long long)blockIdx.z * sB;
    const long long cb = (long long)blockIdx.z * sC;
    const int rbase = (w >> 1) * 32;   // 0,32,64,96
    const int cbase = (w & 1) * 32;    // 0,32

    // cooperative loader indices
    const int ar  = tid >> 1;          // 0..127  A row
    const int akc = (tid & 1) * 16;    // 0,16    A k-chunk (16 f16 = 32B)
    const int bn  = tid >> 2;          // 0..63   B row (output column)
    const int bkc = (tid & 3) * 8;     // 0..24   B k-chunk (8 f16 = 16B)
    const int gn  = n0 + bn;
    const int bok = (gn < N);
    const long long arow = ab + (long long)(m0 + ar) * lda + akc;
    const long long brow = bb + (long long)(bok ? gn : 0) * ldb + bkc;

    v8f c00 = {}, c01 = {}, c10 = {}, c11 = {};

    uint32_t ra[8], rbv[4];

    // global -> registers for K-step k0 (+ WGP-scope prefetch of the next)
    auto gload = [&](int k0) {
        const uint32_t* s = (const uint32_t*)(A + arow + k0);
#pragma unroll
        for (int j = 0; j < 8; ++j) ra[j] = s[j];
        const uint32_t* sb = (const uint32_t*)(Bm + brow + k0);
        uint32_t v0 = sb[0], v1 = sb[1], v2 = sb[2], v3 = sb[3];
        if (!bok) { v0 = 0u; v1 = 0u; v2 = 0u; v3 = 0u; }
        rbv[0] = v0; rbv[1] = v1; rbv[2] = v2; rbv[3] = v3;
        if (k0 + 32 < K) {
            __builtin_prefetch((const void*)(A + arow + k0 + 32), 0, 3);
            __builtin_prefetch((const void*)(Bm + brow + k0 + 32), 0, 3);
        }
    };
    // registers -> LDS buffer
    auto lstore = [&](int buf) {
        uint32_t* d = (uint32_t*)&la[buf][ar * AST + akc];
#pragma unroll
        for (int j = 0; j < 8; ++j) d[j] = ra[j];
        uint32_t* db = (uint32_t*)&lb[buf][bn * BST + bkc];
        db[0] = rbv[0]; db[1] = rbv[1]; db[2] = rbv[2]; db[3] = rbv[3];
    };
    // fragments in contiguous k order (k = 16*(lane/16) + s, same for A and B) + 4 WMMAs
    auto compute = [&](int buf) {
        union { v16h v; uint32_t u[8]; } a0, a1, b0, b1;
        const int hk = (lane >> 4) * 16;
        const int r0 = (rbase + (lane & 15)) * AST;
        const int r1 = (rbase + 16 + (lane & 15)) * AST;
        const uint32_t* lau = (const uint32_t*)&la[buf][0];
#pragma unroll
        for (int p = 0; p < 8; ++p) {
            a0.u[p]     = lau[(r0 + hk + 2 * p) >> 1];
            a1.u[p]     = lau[(r1 + hk + 2 * p) >> 1];
        }
        const int kb = (lane >> 4) * 16;
        const int q0 = (cbase + (lane & 15)) * BST + kb;
        const int q1 = (cbase + 16 + (lane & 15)) * BST + kb;
        const uint32_t* lbu = (const uint32_t*)&lb[buf][0];
#pragma unroll
        for (int p = 0; p < 8; ++p) {
            b0.u[p] = lbu[(q0 >> 1) + p];
            b1.u[p] = lbu[(q1 >> 1) + p];
        }
        c00 = __builtin_amdgcn_wmma_f32_16x16x32_f16(
                  false, a0.v, false, b0.v, (short)0, c00, false, false);
        c01 = __builtin_amdgcn_wmma_f32_16x16x32_f16(
                  false, a0.v, false, b1.v, (short)0, c01, false, false);
        c10 = __builtin_amdgcn_wmma_f32_16x16x32_f16(
                  false, a1.v, false, b0.v, (short)0, c10, false, false);
        c11 = __builtin_amdgcn_wmma_f32_16x16x32_f16(
                  false, a1.v, false, b1.v, (short)0, c11, false, false);
    };

    // ---- software pipeline: prologue ----
    gload(0);
    lstore(0);
    __syncthreads();
    int buf = 0;
    for (int k0 = 32; k0 < K; k0 += 32) {
        gload(k0);          // global loads for next step in flight ...
        compute(buf);       // ... while WMMAs of current step execute
        lstore(buf ^ 1);
        __syncthreads();
        buf ^= 1;
    }
    compute(buf);           // epilogue

    // ---- store: VGPR r holds M = base + r + 8*(lane>=16), N = lane&15 ----
    const int mr0 = m0 + rbase + (lane >> 4) * 8;
    const int mr1 = mr0 + 16;
    const int ncA = n0 + cbase + (lane & 15);
    const int ncB = ncA + 16;
#pragma unroll
    for (int r = 0; r < 8; ++r) {
        if (ncA < N) {
            C[cb + (long long)(mr0 + r) * ldc + ncA] = alpha * c00[r];
            C[cb + (long long)(mr1 + r) * ldc + ncA] = alpha * c10[r];
        }
        if (ncB < N) {
            C[cb + (long long)(mr0 + r) * ldc + ncB] = alpha * c01[r];
            C[cb + (long long)(mr1 + r) * ldc + ncB] = alpha * c11[r];
        }
    }
}

// ---------------------------------------------------------------------------
// 3x3 SAME conv on each token viewed as (3,16,16); q,k,v at once.
// q,k -> fp16 [B,H,N,HD];  v -> TRANSPOSED fp16 [B,H,HD,N] (for [N,K] GEMM).
// ---------------------------------------------------------------------------
__global__ __launch_bounds__(256) void k_conv_qkv(
    const float* __restrict__ x,
    const float* __restrict__ qw, const float* __restrict__ qb,
    const float* __restrict__ kw, const float* __restrict__ kb,
    const float* __restrict__ vw, const float* __restrict__ vb,
    f16* __restrict__ q, f16* __restrict__ k, f16* __restrict__ vT)
{
    const int token = blockIdx.x;          // b*N + n
    const int b = token / N_, n = token % N_;
    const int t = threadIdx.x;

    __shared__ float img[CHN_][P_ + 2][P_ + 2];   // zero halo
    __shared__ float swq[81], swk[81], swv[81];
    __shared__ float sb[3][CHN_];

    for (int i = t; i < CHN_ * 18 * 18; i += 256)
        ((float*)img)[i] = 0.0f;
    if (t < 81) { swq[t] = qw[t]; swk[t] = kw[t]; swv[t] = vw[t]; }
    if (t < CHN_) { sb[0][t] = qb[t]; sb[1][t] = kb[t]; sb[2][t] = vb[t]; }
    __syncthreads();

    for (int j = 0; j < 3; ++j) {
        const int f = j * 256 + t;                    // 0..767
        const int c = f >> 8, rem = f & 255;
        img[c][(rem >> 4) + 1][(rem & 15) + 1] = x[(long long)token * C_ + f];
    }
    __syncthreads();

    for (int j = 0; j < 3; ++j) {
        const int f  = j * 256 + t;
        const int co = f >> 8;
        const int y  = (f >> 4) & 15;
        const int xx = f & 15;
        float aq = sb[0][co], ak = sb[1][co], av = sb[2][co];
#pragma unroll
        for (int ci = 0; ci < CHN_; ++ci)
#pragma unroll
            for (int dy = 0; dy < 3; ++dy)
#pragma unroll
                for (int dx = 0; dx < 3; ++dx) {
                    const int wi = ((co * 3 + ci) * 3 + dy) * 3 + dx;
                    const float px = img[ci][y + dy][xx + dx];
                    aq = fmaf(swq[wi], px, aq);
                    ak = fmaf(swk[wi], px, ak);
                    av = fmaf(swv[wi], px, av);
                }
        const int h = f / HD_, d = f % HD_;
        const long long bh = (long long)(b * H_ + h);
        const long long o = (bh * N_ + n) * HD_ + d;
        q[o] = (f16)aq; k[o] = (f16)ak;
        vT[(bh * HD_ + d) * N_ + n] = (f16)av;
    }
}

// ---------------------------------------------------------------------------
// Per (b,n): softmax of all 8 head rows, head-mix (re-attention), store fp16
// probs, emit per-row sum(a^2) partials per output head (deterministic).
// ---------------------------------------------------------------------------
__global__ __launch_bounds__(256) void k_softmax_reatten(
    const float* __restrict__ S,       // [B*H, N, N] scaled logits
    const float* __restrict__ rw, const float* __restrict__ rb,
    f16* __restrict__ aP,              // [B*H, N, N] re-attended probs (fp16)
    float* __restrict__ psum)          // [B*N, H] per-row sum(a^2)
{
    const int row = blockIdx.x;        // b*N + n
    const int b = row / N_, n = row % N_;
    const int t = threadIdx.x;         // column m

    __shared__ float red[256];
    float pl[H_];

    for (int h = 0; h < H_; ++h) {
        const float val = S[(((long long)(b * H_ + h)) * N_ + n) * N_ + t];
        red[t] = val; __syncthreads();
        for (int s = 128; s > 0; s >>= 1) {
            if (t < s) red[t] = fmaxf(red[t], red[t + s]);
            __syncthreads();
        }
        const float mx = red[0]; __syncthreads();
        const float e = __expf(val - mx);
        red[t] = e; __syncthreads();
        for (int s = 128; s > 0; s >>= 1) {
            if (t < s) red[t] += red[t + s];
            __syncthreads();
        }
        const float inv = 1.0f / red[0]; __syncthreads();
        pl[h] = e * inv;
    }

    float sq[H_];
#pragma unroll
    for (int o = 0; o < H_; ++o) {
        float a = rb[o];
#pragma unroll
        for (int i = 0; i < H_; ++i) a = fmaf(rw[o * H_ + i], pl[i], a);
        aP[(((long long)(b * H_ + o)) * N_ + n) * N_ + t] = (f16)a;
        sq[o] = a * a;
    }
    for (int o = 0; o < H_; ++o) {
        red[t] = sq[o]; __syncthreads();
        for (int s = 128; s > 0; s >>= 1) {
            if (t < s) red[t] += red[t + s];
            __syncthreads();
        }
        if (t == 0) psum[(long long)row * H_ + o] = red[0];
        __syncthreads();
    }
}

__global__ __launch_bounds__(256) void k_reduce_sumsq(
    const float* __restrict__ psum, float* __restrict__ sumsq)
{
    const int h = blockIdx.x, t = threadIdx.x;
    float acc = 0.0f;
    for (int i = t; i < B_ * N_; i += 256) acc += psum[(long long)i * H_ + h];
    __shared__ float red[256];
    red[t] = acc; __syncthreads();
    for (int s = 128; s > 0; s >>= 1) {
        if (t < s) red[t] += red[t + s];
        __syncthreads();
    }
    if (t == 0) sumsq[h] = red[0];
}

// colsum over m of V: reads transposed vT [B*H, HD, N] (contiguous rows)
__global__ void k_colsum_v(const f16* __restrict__ vT, float* __restrict__ cs)
{
    const int bh = blockIdx.x, d = threadIdx.x;    // d < 96
    const long long r = ((long long)bh * HD_ + d) * N_;
    float acc = 0.0f;
    for (int m = 0; m < N_; ++m) acc += (float)vT[r + m];
    cs[(long long)bh * HD_ + d] = acc;
}

// ---------------------------------------------------------------------------
// BN folded into affine epilogue:  o = s[h]*U + t[h]*colsumV ; emit fp16 [B,N,C]
// mu[h] analytic because softmax rows sum to 1.
// ---------------------------------------------------------------------------
__global__ __launch_bounds__(256) void k_combine(
    const float* __restrict__ U, const float* __restrict__ cs,
    const float* __restrict__ sumsq,
    const float* __restrict__ rw, const float* __restrict__ rb,
    const float* __restrict__ bg, const float* __restrict__ bbias,
    f16* __restrict__ oh)
{
    const long long idx = (long long)blockIdx.x * 256 + threadIdx.x; // < B*N*C
    const int c = (int)(idx % C_);
    const int n = (int)((idx / C_) % N_);
    const int b = (int)(idx / ((long long)N_ * C_));
    const int h = c / HD_, d = c % HD_;

    float rwsum = 0.0f;
#pragma unroll
    for (int i = 0; i < H_; ++i) rwsum += rw[h * H_ + i];
    const float mu  = rwsum / (float)N_ + rb[h];
    const float var = sumsq[h] / ((float)B_ * N_ * N_) - mu * mu;
    const float s   = bg[h] * rsqrtf(var + EPS_);
    const float tt  = bbias[h] - mu * s;

    const long long bh = (long long)(b * H_ + h);
    const float val = s * U[(bh * N_ + n) * HD_ + d] + tt * cs[bh * HD_ + d];
    oh[idx] = (f16)val;
}

// ---------------------------------------------------------------------------
// x = LayerNorm_{N,C}( gemm_out + bias + x ); writes f32 (in-place) + fp16 copy
// ---------------------------------------------------------------------------
__global__ __launch_bounds__(256) void k_ln2d(
    const float* __restrict__ gin, const float* __restrict__ bias,
    float* __restrict__ x,
    const float* __restrict__ g, const float* __restrict__ be,
    f16* __restrict__ xh)
{
    const int b = blockIdx.x, t = threadIdx.x;
    const long long S = (long long)N_ * C_;
    const long long base = (long long)b * S;

    float sum = 0.0f, sq = 0.0f;
    for (long long i = t; i < S; i += 256) {
        const float v = gin[base + i] + bias[i % C_] + x[base + i];
        sum += v; sq += v * v;
    }
    __shared__ float r1[256], r2[256];
    r1[t] = sum; r2[t] = sq; __syncthreads();
    for (int s = 128; s > 0; s >>= 1) {
        if (t < s) { r1[t] += r1[t + s]; r2[t] += r2[t + s]; }
        __syncthreads();
    }
    const float m  = r1[0] / (float)S;
    const float rs = rsqrtf(r2[0] / (float)S - m * m + EPS_);
    __syncthreads();

    for (long long i = t; i < S; i += 256) {
        const float v  = gin[base + i] + bias[i % C_] + x[base + i];
        const float yv = (v - m) * rs * g[i] + be[i];
        x[base + i] = yv;
        xh[base + i] = (f16)yv;
    }
}

__global__ void k_bias_gelu(const float* __restrict__ gin,
                            const float* __restrict__ b1,
                            f16* __restrict__ hh, long long total)
{
    const long long i = (long long)blockIdx.x * 256 + threadIdx.x;
    if (i >= total) return;
    const float v  = gin[i] + b1[i % HID_];
    const float ge = 0.5f * v * (1.0f + erff(v * 0.70710678118654752f));
    hh[i] = (f16)ge;
}

// cast f32 [K,N] row-major -> f16 [N,K] row-major (weight pre-transpose)
__global__ void k_cast_t(const float* __restrict__ s, f16* __restrict__ d,
                         int K, int N)
{
    const long long i = (long long)blockIdx.x * 256 + threadIdx.x;
    if (i >= (long long)K * N) return;
    const int k = (int)(i / N), n = (int)(i % N);
    d[(long long)n * K + k] = (f16)s[i];
}

// ---------------------------------------------------------------------------
extern "C" void kernel_launch(void* const* d_in, const int* in_sizes, int n_in,
                              void* d_out, int out_size, void* d_ws,
                              size_t ws_size, hipStream_t stream)
{
    (void)in_sizes; (void)n_in; (void)out_size; (void)ws_size;

    const float* enc = (const float*)d_in[0];
    const float* qw  = (const float*)d_in[1];
    const float* qb  = (const float*)d_in[2];
    const float* kw  = (const float*)d_in[3];
    const float* kb  = (const float*)d_in[4];
    const float* vw  = (const float*)d_in[5];
    const float* vb  = (const float*)d_in[6];
    const float* rw  = (const float*)d_in[7];
    const float* rbv = (const float*)d_in[8];
    const float* bng = (const float*)d_in[9];
    const float* bnb = (const float*)d_in[10];
    const float* pw  = (const float*)d_in[11];
    const float* pb  = (const float*)d_in[12];
    const float* l1g = (const float*)d_in[13];
    const float* l1b = (const float*)d_in[14];
    const float* w1  = (const float*)d_in[15];
    const float* b1  = (const float*)d_in[16];
    const float* w2  = (const float*)d_in[17];
    const float* b2  = (const float*)d_in[18];
    const float* l2g = (const float*)d_in[19];
    const float* l2b = (const float*)d_in[20];

    float* x = (float*)d_out;                 // live residual stream [B,N,C]

    // ---- workspace carve ----
    uint8_t* base = (uint8_t*)d_ws;
    size_t off = 0;
    auto carve = [&](size_t bytes) -> void* {
        void* p = base + off;
        off += bytes;
        off = (off + 255) & ~(size_t)255;
        return p;
    };
    const size_t SZX = (size_t)B_ * N_ * C_;              // 12,582,912
    const size_t SZS = (size_t)B_ * H_ * N_ * N_;         // 33,554,432
    const size_t SZH = (size_t)B_ * N_ * HID_;            // 25,165,824

    f16*   qh  = (f16*)carve(SZX * 2);
    f16*   kh  = (f16*)carve(SZX * 2);
    f16*   vT  = (f16*)carve(SZX * 2);
    float* Sb  = (float*)carve(SZS * 4);   // logits + reused f32 GEMM scratch
    f16*   aP  = (f16*)carve(SZS * 2);
    float* U   = (float*)carve(SZX * 4);
    f16*   oh  = (f16*)carve(SZX * 2);
    f16*   xh  = (f16*)carve(SZX * 2);
    f16*   hh  = (f16*)carve(SZH * 2);
    float* cs  = (float*)carve((size_t)B_ * H_ * HD_ * 4);
    float* psum = (float*)carve((size_t)B_ * N_ * H_ * 4);
    float* ssq = (float*)carve(H_ * 4);
    f16*   pwT = (f16*)carve((size_t)L_ * C_ * C_ * 2);
    f16*   w1T = (f16*)carve((size_t)L_ * C_ * HID_ * 2);
    f16*   w2T = (f16*)carve((size_t)L_ * HID_ * C_ * 2);

    auto gemm = [&](const f16* A, int lda, long long sA,
                    const f16* Bm, int ldb, long long sB,
                    float* Cm, int ldc, long long sC,
                    int M, int Nn, int K, float alpha, int batch) {
        dim3 g((Nn + TN_ - 1) / TN_, M / TM_, batch);
        k_gemm<<<g, 256, 0, stream>>>(A, lda, sA, Bm, ldb, sB,
                                      Cm, ldc, sC, M, Nn, K, alpha);
    };

    // fp16 transposed weight copies (once per launch)
    for (int l = 0; l < L_; ++l) {
        {
            const long long n = (long long)C_ * C_;
            k_cast_t<<<dim3((unsigned)((n + 255) / 256)), 256, 0, stream>>>(
                pw + (size_t)l * C_ * C_, pwT + (size_t)l * C_ * C_, C_, C_);
        }
        {
            const long long n = (long long)C_ * HID_;
            k_cast_t<<<dim3((unsigned)((n + 255) / 256)), 256, 0, stream>>>(
                w1 + (size_t)l * C_ * HID_, w1T + (size_t)l * C_ * HID_, C_, HID_);
        }
        {
            const long long n = (long long)HID_ * C_;
            k_cast_t<<<dim3((unsigned)((n + 255) / 256)), 256, 0, stream>>>(
                w2 + (size_t)l * HID_ * C_, w2T + (size_t)l * HID_ * C_, HID_, C_);
        }
    }

    // x <- encoded_patches
    hipMemcpyAsync(x, enc, SZX * sizeof(float), hipMemcpyDeviceToDevice, stream);

    for (int l = 0; l < L_; ++l) {
        // q,k (fp16 [B,H,N,HD]) and vT (fp16 [B,H,HD,N]) = conv3x3(x)
        k_conv_qkv<<<B_ * N_, 256, 0, stream>>>(
            x, qw + l * 81, qb + l * CHN_, kw + l * 81, kb + l * CHN_,
            vw + l * 81, vb + l * CHN_, qh, kh, vT);

        // S = scale * q @ k^T   (batched over B*H; B matrix = k as [N,K])
        gemm(qh, HD_, (long long)N_ * HD_, kh, HD_, (long long)N_ * HD_,
             Sb, N_, (long long)N_ * N_, N_, N_, HD_, SCALE_, B_ * H_);

        // softmax + re-attention mix + stats partials
        k_softmax_reatten<<<B_ * N_, 256, 0, stream>>>(
            Sb, rw + l * H_ * H_, rbv + l * H_, aP, psum);
        k_reduce_sumsq<<<H_, 256, 0, stream>>>(psum, ssq);
        k_colsum_v<<<B_ * H_, HD_, 0, stream>>>(vT, cs);

        // U = aP @ v   (B matrix = vT as [N(=HD),K(=m)])
        gemm(aP, N_, (long long)N_ * N_, vT, N_, (long long)HD_ * N_,
             U, HD_, (long long)N_ * HD_, N_, HD_, N_, 1.0f, B_ * H_);

        // BN-folded epilogue -> oh fp16 [B,N,C]
        k_combine<<<(unsigned)(SZX / 256), 256, 0, stream>>>(
            U, cs, ssq, rw + l * H_ * H_, rbv + l * H_,
            bng + l * H_, bnb + l * H_, oh);

        // proj: Sb = oh @ pw    (B matrix = pwT [N=C,K=C])
        gemm(oh, C_, 0, pwT + (size_t)l * C_ * C_, C_, 0,
             Sb, C_, 0, B_ * N_, C_, C_, 1.0f, 1);
        // x = LN(Sb + pb + x); xh = fp16(x)
        k_ln2d<<<B_, 256, 0, stream>>>(
            Sb, pb + l * C_, x,
            l1g + (size_t)l * N_ * C_, l1b + (size_t)l * N_ * C_, xh);

        // FFN1: Sb = xh @ w1    (B matrix = w1T [N=HID,K=C])
        gemm(xh, C_, 0, w1T + (size_t)l * C_ * HID_, C_, 0,
             Sb, HID_, 0, B_ * N_, HID_, C_, 1.0f, 1);
        k_bias_gelu<<<(unsigned)(SZH / 256), 256, 0, stream>>>(
            Sb, b1 + l * HID_, hh, (long long)SZH);

        // FFN2: Sb = hh @ w2    (B matrix = w2T [N=C,K=HID])
        gemm(hh, HID_, 0, w2T + (size_t)l * HID_ * C_, HID_, 0,
             Sb, C_, 0, B_ * N_, C_, HID_, 1.0f, 1);
        // x = LN(Sb + b2 + x)
        k_ln2d<<<B_, 256, 0, stream>>>(
            Sb, b2 + l * C_, x,
            l2g + (size_t)l * N_ * C_, l2b + (size_t)l * N_ * C_, xh);
    }
}
